// ContextTransformCTRNN_45896020525286
// MI455X (gfx1250) — matrix operand order repaired and checked by reading, not compiled, not run on hardware
//
#include <hip/hip_runtime.h>
#include <hip/hip_bf16.h>

// ============================================================================
// ContextTransformCTRNN for MI455X (gfx1250, wave32, WMMA + TDM).
//
// Key algebraic facts exploited (faithful to the reference):
//  * `hidden = transformed` overwrites the leaky hidden update -> w_h2h/b_h2h
//    and the relu'd h_new are DEAD CODE. output_t == transformed_t.
//  * context recurrence is independent of hidden -> run it once sequentially
//    (cheap), then the dominant hypernet contraction (34.4 GMAC total) is
//    fully parallel over (t, i) and runs as bf16 WMMA out of L2
//    (w_c2t = 64MB fp32 / 32MB bf16, resident in the 192MB L2).
//  * num_steps == 1 in the harness.
//
// Data movement: K3 stages its three operand blocks (W tile 128KB, ctx^T 16KB,
// ip 32KB) with TENSOR_LOAD_TO_LDS, using TDM pad_interval/pad_amount to
// produce the bank-conflict-avoiding padded LDS row strides directly.
//
// Workspace layout (requires ws_size >= 37 MB):
//   [ 0MB..32MB)  Wb   : w_c2t converted to bf16            (65536*256)
//   [32MB..33MB)  CTXb : bf16 context entering step t, [t][k][b] (64*256*32)
//   [33MB..35MB)  IP   : fp32 x@w_i2h^T + b_i2h, [t][b][j]  (64*32*256)
//   [35MB..37MB)  XC   : fp32 x@w_i2c^T + b_i2c, [t][b][h]  (64*32*256)
// ============================================================================

typedef __attribute__((ext_vector_type(16))) __bf16 v16bf;
typedef __attribute__((ext_vector_type(8)))  float  v8f;
typedef unsigned int u32x4 __attribute__((ext_vector_type(4)));
typedef int          i32x4 __attribute__((ext_vector_type(4)));
typedef int          i32x8 __attribute__((ext_vector_type(8)));

#define T_STEPS 64
#define B_SZ    32
#define I_SZ    128
#define H_SZ    256
#define TBH     (T_STEPS * B_SZ * H_SZ)   // 524288
#define BH      (B_SZ * H_SZ)             // 8192
#define ALPHA_F 0.2f

// LDS row strides (elements), padded so 16-lane fragment reads hit distinct
// banks (stride % 64 words == 4) while staying 16B-aligned for b128 access.
#define AS 264   // bf16 elems per W-tile row   (256 + 8)  -> 528 B
#define CS 40    // bf16 elems per ctx^T row    (32 + 8)   -> 80 B
#define IS 260   // f32  elems per ip row       (256 + 4)  -> 1040 B

#define K2_SMEM ((256*AS + 32*AS) * 2 + 32*IS * 4 + 256 * 4)            // 186368 B
#define K3_SMEM (256*AS*2 + 256*CS*2 + 32*IS*4 + 256*4 + 32*4)          // 190080 B

// ----------------------------------------------------------------------------
// Tensor Data Mover: 1D contiguous global -> LDS with interval padding.
// D# per CDNA5 ISA:
//   group0: [1:0]=count=1, [63:32]=lds_addr, [120:64]=global_addr, [127:126]=2
//   group1: [17:16]=data_size(2 -> 4B), [20]=pad_enable, [24:22]=pad_interval,
//           [31:25]=pad_amount, [79:48]=tensor_dim0, [111:80]=tensor_dim1=1,
//           [127:112]=tile_dim0, [143:128]=tile_dim1=1, [207:160]=dim0_stride
//   groups 2/3: zero (<=2D tensor)
// pad_interval code p: pad after 2^(p+1) dwords; pad_amount code a: (a+1) dwords.
// ----------------------------------------------------------------------------
#if __has_include(<hip/amd_detail/amd_gfx1250_TDM.h>)
#define TDM_CALL(g0, g1, z4) do { i32x8 _z8 = {0,0,0,0,0,0,0,0};                 \
    __builtin_amdgcn_tensor_load_to_lds((g0), (g1), (z4), (z4), _z8, 0); } while (0)
#else
#define TDM_CALL(g0, g1, z4) __builtin_amdgcn_tensor_load_to_lds((g0), (g1), (z4), (z4), 0)
#endif

__device__ __forceinline__ void tdm_load_1d_padded(const void* lds_dst, const void* gsrc,
                                                   unsigned n4 /* 4-byte units, <= 65535 */,
                                                   unsigned pad_int, unsigned pad_amt) {
  unsigned long long ga = (unsigned long long)(size_t)gsrc;
  unsigned lds_off = (unsigned)(size_t)lds_dst;            // generic addr low 32 = LDS offset
  u32x4 g0 = { 1u, lds_off, (unsigned)ga, (unsigned)(ga >> 32) | (2u << 30) };
  i32x8 g1 = { 0, 0, 0, 0, 0, 0, 0, 0 };
  g1[0] = (int)((2u << 16) | (1u << 20) | (pad_int << 22) | (pad_amt << 25));
  g1[1] = (int)(n4 << 16);                                 // tensor_dim0[15:0]
  g1[2] = (int)((n4 >> 16) | (1u << 16));                  // tensor_dim0[31:16], tensor_dim1=1
  g1[3] = (int)(n4 << 16);                                 // tile_dim0 = n4
  g1[4] = 1;                                               // tile_dim1 = 1
  g1[5] = (int)n4;                                         // tensor_dim0_stride[31:0]
  i32x4 z4 = { 0, 0, 0, 0 };
  TDM_CALL(g0, g1, z4);
}

// ----------------------------------------------------------------------------
// K0: w_c2t fp32 -> bf16 (one-time; halves the per-step L2 stream).
// ----------------------------------------------------------------------------
__global__ void convert_w_kernel(const float* __restrict__ w, __bf16* __restrict__ out) {
  int idx = blockIdx.x * blockDim.x + threadIdx.x;        // over 4,194,304 float4s
  float4 v = reinterpret_cast<const float4*>(w)[idx];
  alignas(8) __bf16 tmp[4] = { (__bf16)v.x, (__bf16)v.y, (__bf16)v.z, (__bf16)v.w };
  *reinterpret_cast<uint2*>(out + 4 * (size_t)idx) = *reinterpret_cast<const uint2*>(tmp);
}

// ----------------------------------------------------------------------------
// K1: batch projections for all timesteps (no recurrence):
//     IP[t,b,h] = x_t @ w_i2h^T + b_i2h ; XC[t,b,h] = x_t @ w_i2c^T + b_i2c
// 134 MMAC total -> plain VALU dot products, fully coalesced.
// ----------------------------------------------------------------------------
__global__ void proj_kernel(const float* __restrict__ x,
                            const float* __restrict__ w_i2h, const float* __restrict__ b_i2h,
                            const float* __restrict__ w_i2c, const float* __restrict__ b_i2c,
                            float* __restrict__ IP, float* __restrict__ XC) {
  int idx = blockIdx.x * blockDim.x + threadIdx.x;        // over 2*TBH
  int which = (idx >= TBH) ? 1 : 0;                       // block-uniform (TBH % 256 == 0)
  int r = idx - which * TBH;
  int h = r & (H_SZ - 1);
  const float4* xr = reinterpret_cast<const float4*>(x + (size_t)(r >> 8) * I_SZ);
  const float4* wr = reinterpret_cast<const float4*>((which ? w_i2c : w_i2h) + (size_t)h * I_SZ);
  float s = (which ? b_i2c : b_i2h)[h];
  #pragma unroll
  for (int q = 0; q < I_SZ / 4; ++q) {
    float4 a = xr[q], w4 = wr[q];
    s += a.x * w4.x + a.y * w4.y + a.z * w4.z + a.w * w4.w;
  }
  (which ? XC : IP)[r] = s;
}

// ----------------------------------------------------------------------------
// K2: sequential context recurrence, single 1024-thread workgroup (32 waves).
//     Per step: 256x WMMA bf16 for c@w_c2c^T, relu + leaky blend in fp32 LDS.
//     Emits CTXb[t][k][b] (bf16, k-major == transposed) for K3's B-matrix,
//     and the final fp32 context into d_out tail.
// ----------------------------------------------------------------------------
__global__ void context_kernel(const float* __restrict__ w_c2c, const float* __restrict__ b_c2c,
                               const float* __restrict__ XC, __bf16* __restrict__ CTXb,
                               float* __restrict__ d_out) {
  extern __shared__ char smem[];
  __bf16* W_s   = reinterpret_cast<__bf16*>(smem);              // [k][h] transposed, 256*AS
  float*  ctx_s = reinterpret_cast<float*>(W_s + 256 * AS);     // [b][h] fp32, 32*IS
  __bf16* cb_s  = reinterpret_cast<__bf16*>(ctx_s + 32 * IS);   // [b][k] bf16, 32*AS
  float*  bcc_s = reinterpret_cast<float*>(cb_s + 32 * AS);     // 256

  const int tid  = threadIdx.x;
  const int lane = tid & 31, wid = tid >> 5;

  // Stage w_c2c^T as bf16 (reads coalesced over k).
  #pragma unroll 4
  for (int c = 0; c < 64; ++c) {
    int idx = c * 1024 + tid;
    int h = idx >> 8, k = idx & 255;
    W_s[k * AS + h] = (__bf16)w_c2c[idx];
  }
  if (tid < H_SZ) bcc_s[tid] = b_c2c[tid];
  #pragma unroll
  for (int c = 0; c < 8; ++c) {                                  // ctx_0 = 0
    int idx = c * 1024 + tid;
    ctx_s[(idx >> 8) * IS + (idx & 255)] = 0.f;
  }
  __syncthreads();

  const int mi = wid & 1, nt = wid >> 1;                         // M-tile (b), N-tile (h)
  const int half = lane >> 4, ln = lane & 15;
  const int h = nt * 16 + ln;

  for (int t = 0; t < T_STEPS; ++t) {
    // Snapshot ENTRY context: bf16 copies for this step's A-matrix + K3's B.
    #pragma unroll
    for (int c = 0; c < 8; ++c) {
      int idx = c * 1024 + tid;
      int b = idx >> 8, k = idx & 255;
      __bf16 bf = (__bf16)ctx_s[b * IS + k];
      cb_s[b * AS + k] = bf;
      CTXb[(size_t)t * BH + k * B_SZ + b] = bf;                  // [t][k][b]
    }
    __syncthreads();

    // acc init: bias + input-context projection (fold the adds into C).
    v8f acc;
    #pragma unroll
    for (int r = 0; r < 8; ++r) {
      int b = mi * 16 + 8 * half + r;
      acc[r] = bcc_s[h] + XC[(size_t)t * BH + b * H_SZ + h];
    }
    // GEMM: [32b x 256k] @ [256k x 256h], K tiled by 32.
    #pragma unroll
    for (int kc = 0; kc < 8; ++kc) {
      int k0 = kc * 32;
      v16bf a, bm;
      { const __bf16* p = cb_s + (mi * 16 + ln) * AS + k0 + 8 * half;   // A: lane=M row
        reinterpret_cast<uint4*>(&a)[0] = *reinterpret_cast<const uint4*>(p);
        reinterpret_cast<uint4*>(&a)[1] = *reinterpret_cast<const uint4*>(p + 16); }
      { const __bf16* q = W_s + k0 * AS + nt * 16 + ln;                 // B: lane%16=N col; K split by lane-half
        #pragma unroll
        for (int s = 0; s < 16; ++s) {
          int kk = (s < 8) ? (8 * half + s) : (16 + 8 * half + (s - 8));
          bm[s] = q[kk * AS];
        } }
      acc = __builtin_amdgcn_wmma_f32_16x16x32_bf16(false, a, false, bm,
                                                    (short)0, acc, false, false);
    }
    // relu + leaky blend back into fp32 context.
    #pragma unroll
    for (int r = 0; r < 8; ++r) {
      int b = mi * 16 + 8 * half + r;
      float nv = fmaxf(acc[r], 0.f);
      float* pc = &ctx_s[b * IS + h];
      *pc = (*pc) * (1.f - ALPHA_F) + ALPHA_F * nv;
    }
    __syncthreads();
  }

  // Final context (post-update at t=63) -> d_out tail.
  #pragma unroll
  for (int c = 0; c < 8; ++c) {
    int idx = c * 1024 + tid;
    d_out[TBH + BH + idx] = ctx_s[(idx >> 8) * IS + (idx & 255)];
  }
}

// ----------------------------------------------------------------------------
// K3: the hypernet contraction (34.4 GMAC). One workgroup per (i, t):
//   tm[j,b] = sum_k Wb[i*256+j, k]*ctx_t[b,k] + b_c2t[i*256+j]   (WMMA, fp32 acc)
//   out[t,b,i] = sum_j tm[j,b] * ip_t[b,j]                       (epilogue)
// The three operand blocks are DMA'd into padded LDS by the Tensor Data Mover:
// wave0 -> W block (128KB contiguous), wave1 -> ctx^T (16KB), wave2 -> ip (32KB),
// each wave s_wait_tensorcnt(0) then the workgroup barrier publishes the LDS.
// ----------------------------------------------------------------------------
__global__ void hyper_kernel(const __bf16* __restrict__ Wb, const __bf16* __restrict__ CTXb,
                             const float* __restrict__ IP, const float* __restrict__ b_c2t,
                             float* __restrict__ d_out) {
  extern __shared__ char smem[];
  __bf16* A_s    = reinterpret_cast<__bf16*>(smem);               // [j][k], 256*AS
  __bf16* C_s    = A_s + 256 * AS;                                // [k][b], 256*CS
  float*  ip_s   = reinterpret_cast<float*>(C_s + 256 * CS);      // [b][j], 32*IS
  float*  bias_s = ip_s + 32 * IS;                                // 256
  float*  out_s  = bias_s + 256;                                  // 32

  const int i = blockIdx.x, t = blockIdx.y;
  const int tid = threadIdx.x, lane = tid & 31, wid = tid >> 5;

  // --- TDM staging (per-wave issue; EXEC ignored by tensor ops) -------------
  if (wid == 0) {
    // 128KB W block: 32768 dwords, pad 4 dwords every 128 -> row stride 528B.
    tdm_load_1d_padded(A_s, Wb + (size_t)i * (H_SZ * H_SZ), 32768u, 6u, 3u);
  } else if (wid == 1) {
    // 16KB ctx^T: 4096 dwords, pad 4 dwords every 16 -> row stride 80B.
    tdm_load_1d_padded(C_s, CTXb + (size_t)t * BH, 4096u, 3u, 3u);
  } else if (wid == 2) {
    // 32KB ip: 8192 dwords, pad 4 dwords every 256 -> row stride 1040B.
    tdm_load_1d_padded(ip_s, IP + (size_t)t * BH, 8192u, 7u, 3u);
  }
  if (wid < 3) __builtin_amdgcn_s_wait_tensorcnt(0);
  if (tid < H_SZ) bias_s[tid] = b_c2t[(size_t)i * H_SZ + tid];
  if (tid < B_SZ) out_s[tid] = 0.f;
  __syncthreads();

  const int j0 = wid * 32;                   // this wave owns j rows [j0, j0+32)
  const int half = lane >> 4, ln = lane & 15;

  v8f acc[2][2];                             // [m-tile][n-tile], bias folded into C
  #pragma unroll
  for (int mi = 0; mi < 2; ++mi) {
    #pragma unroll
    for (int r = 0; r < 8; ++r) acc[mi][0][r] = bias_s[j0 + 16 * mi + 8 * half + r];
    acc[mi][1] = acc[mi][0];
  }

  #pragma unroll
  for (int kc = 0; kc < 8; ++kc) {
    int k0 = kc * 32;
    v16bf a[2], b[2];
    #pragma unroll
    for (int mi = 0; mi < 2; ++mi) {         // A: lane%16 = M row; K split by lane-half
      const __bf16* p = A_s + (j0 + 16 * mi + ln) * AS + k0 + 8 * half;
      reinterpret_cast<uint4*>(&a[mi])[0] = *reinterpret_cast<const uint4*>(p);
      reinterpret_cast<uint4*>(&a[mi])[1] = *reinterpret_cast<const uint4*>(p + 16);
    }
    #pragma unroll
    for (int ni = 0; ni < 2; ++ni) {         // B: lane%16 = N col; K split by lane-half
      const __bf16* q = C_s + k0 * CS + 16 * ni + ln;
      #pragma unroll
      for (int s = 0; s < 16; ++s) {
        int kk = (s < 8) ? (8 * half + s) : (16 + 8 * half + (s - 8));
        b[ni][s] = q[kk * CS];
      }
    }
    #pragma unroll
    for (int mi = 0; mi < 2; ++mi)
      #pragma unroll
      for (int ni = 0; ni < 2; ++ni)
        acc[mi][ni] = __builtin_amdgcn_wmma_f32_16x16x32_bf16(false, a[mi], false, b[ni],
                                                              (short)0, acc[mi][ni], false, false);
  }

  // Epilogue: out[b] += sum_j tm[j,b] * ip[b,j]  (C layout: m=r+8*half, n=lane%16)
  #pragma unroll
  for (int ni = 0; ni < 2; ++ni) {
    int b = 16 * ni + ln;
    float part = 0.f;
    #pragma unroll
    for (int mi = 0; mi < 2; ++mi)
      #pragma unroll
      for (int r = 0; r < 8; ++r)
        part += acc[mi][ni][r] * ip_s[b * IS + j0 + 16 * mi + 8 * half + r];
    atomicAdd(&out_s[b], part);              // ds_add_f32, 32 slots
  }
  __syncthreads();

  if (tid < B_SZ) {
    float v = out_s[tid];
    d_out[((size_t)t * B_SZ + tid) * H_SZ + i] = v;         // output[t,b,i]
    if (t == T_STEPS - 1)
      d_out[TBH + (size_t)tid * H_SZ + i] = v;              // hidden == transformed_63
  }
}

// ----------------------------------------------------------------------------
extern "C" void kernel_launch(void* const* d_in, const int* in_sizes, int n_in,
                              void* d_out, int out_size, void* d_ws, size_t ws_size,
                              hipStream_t stream) {
  const float* x     = (const float*)d_in[0];
  const float* w_i2h = (const float*)d_in[1];
  const float* b_i2h = (const float*)d_in[2];
  // d_in[3]=w_h2h, d_in[4]=b_h2h: provably dead in the reference (hidden is
  // overwritten by `transformed` every step). d_in[11]=num_steps==1.
  const float* w_i2c = (const float*)d_in[5];
  const float* b_i2c = (const float*)d_in[6];
  const float* w_c2c = (const float*)d_in[7];
  const float* b_c2c = (const float*)d_in[8];
  const float* w_c2t = (const float*)d_in[9];
  const float* b_c2t = (const float*)d_in[10];
  float* out = (float*)d_out;

  char* ws = (char*)d_ws;                       // needs >= 37 MB
  __bf16* Wb   = (__bf16*)(ws);
  __bf16* CTXb = (__bf16*)(ws + ((size_t)32 << 20));
  float*  IP   = (float*) (ws + ((size_t)33 << 20));
  float*  XC   = (float*) (ws + ((size_t)35 << 20));

  convert_w_kernel<<<16384, 256, 0, stream>>>(w_c2t, Wb);
  proj_kernel<<<(2 * TBH) / 256, 256, 0, stream>>>(x, w_i2h, b_i2h, w_i2c, b_i2c, IP, XC);
  context_kernel<<<1, 1024, K2_SMEM, stream>>>(w_c2c, b_c2c, XC, CTXb, out);
  hyper_kernel<<<dim3(H_SZ, T_STEPS), 256, K3_SMEM, stream>>>(Wb, CTXb, IP, b_c2t, out);
}
